// TensorNet_74947179316248
// MI455X (gfx1250) — compile-verified, matrix operand order checked by reading, not run on hardware
//
#include <hip/hip_runtime.h>
#include <math.h>

#define N_ATOMS 8192
#define E_EDGES 65536
#define H_DIM   128
#define R_RBF   32
#define CUT_R   4.5f

typedef __attribute__((ext_vector_type(16))) _Float16 v16h;
typedef __attribute__((ext_vector_type(8)))  _Float16 v8h;
typedef __attribute__((ext_vector_type(8)))  float    v8f;

__device__ __forceinline__ float cutoff_fn(float d) {
  float c = 0.5f * (__cosf(d * (3.14159265358979323846f / CUT_R)) + 1.0f);
  return (d < CUT_R) ? c : 0.0f;
}
__device__ __forceinline__ float silu_f(float x) { return x / (1.0f + __expf(-x)); }

// Fragment packing per ISA 16-bit layout: lane needs K runs [kb..kb+7] and [kb+16..kb+23].
__device__ __forceinline__ v16h pack_frag(const float* __restrict__ p, int kb) {
  float4 q0 = *(const float4*)(p + kb);
  float4 q1 = *(const float4*)(p + kb + 4);
  float4 q2 = *(const float4*)(p + kb + 16);
  float4 q3 = *(const float4*)(p + kb + 20);
  v16h v;
  v[0]=(_Float16)q0.x; v[1]=(_Float16)q0.y; v[2]=(_Float16)q0.z; v[3]=(_Float16)q0.w;
  v[4]=(_Float16)q1.x; v[5]=(_Float16)q1.y; v[6]=(_Float16)q1.z; v[7]=(_Float16)q1.w;
  v[8]=(_Float16)q2.x; v[9]=(_Float16)q2.y; v[10]=(_Float16)q2.z; v[11]=(_Float16)q2.w;
  v[12]=(_Float16)q3.x; v[13]=(_Float16)q3.y; v[14]=(_Float16)q3.z; v[15]=(_Float16)q3.w;
  return v;
}
__device__ __forceinline__ v16h pack_frag(const _Float16* __restrict__ p, int kb) {
  v8h q0 = *(const v8h*)(p + kb);        // 16B = global_load_b128, no cvt
  v8h q1 = *(const v8h*)(p + kb + 16);
  v16h v;
#pragma unroll
  for (int i = 0; i < 8; ++i) { v[i] = q0[i]; v[8+i] = q1[i]; }
  return v;
}

// full 3x3 from compact [i, a01,a02,a12, s00,s11,s22, s01,s02,s12]
__device__ __forceinline__ void full_from_compact(const float* c, float* T) {
  float i=c[0],a01=c[1],a02=c[2],a12=c[3],s00=c[4],s11=c[5],s22=c[6],s01=c[7],s02=c[8],s12=c[9];
  T[0]=i+s00; T[1]=a01+s01; T[2]=a02+s02;
  T[3]=s01-a01; T[4]=i+s11; T[5]=a12+s12;
  T[6]=s02-a02; T[7]=s12-a12; T[8]=i+s22;
}
__device__ __forceinline__ void compact_from_full(const float* T, float* c) {
  float tr3 = (T[0]+T[4]+T[8]) * (1.0f/3.0f);
  c[0]=tr3;
  c[1]=0.5f*(T[1]-T[3]); c[2]=0.5f*(T[2]-T[6]); c[3]=0.5f*(T[5]-T[7]);
  c[4]=T[0]-tr3; c[5]=T[4]-tr3; c[6]=T[8]-tr3;
  c[7]=0.5f*(T[1]+T[3]); c[8]=0.5f*(T[2]+T[6]); c[9]=0.5f*(T[5]+T[7]);
}
__device__ __forceinline__ void mm3(const float* A, const float* B, float* C) {
#pragma unroll
  for (int i=0;i<3;i++)
#pragma unroll
    for (int j=0;j<3;j++)
      C[i*3+j] = A[i*3+0]*B[0*3+j] + A[i*3+1]*B[1*3+j] + A[i*3+2]*B[2*3+j];
}

// ---------------------------------------------------------------------------
// WMMA GEMM: C[M,Nout] = act(A[M,K] @ Wh[Nout,K]^T + bias) [* cutoff(ew[row])]
// Wave -> 16x32 tile (2 accumulators share one A fragment). Weights pre-converted
// to f16 (Wh). A is f32 or f16 per call site; output f32 or f16 per call site.
// ---------------------------------------------------------------------------
template<typename AT, typename OT, int ACT, int RS>
__global__ __launch_bounds__(256)
void k_gemm(const AT* __restrict__ A, const _Float16* __restrict__ Wh,
            const float* __restrict__ bias, OT* __restrict__ C,
            int M, int Nout, int K, const float* __restrict__ ew)
{
  const int tilesN2 = Nout >> 5;
  const long total = (long)(M >> 4) * tilesN2;
  const int lane = threadIdx.x & 31;
  long t = (long)blockIdx.x * 8 + (threadIdx.x >> 5);
  if (t >= total) return;                       // wave-uniform: EXEC stays all-1s
  const int tn = (int)(t % tilesN2);
  const int rowBase = (int)(t / tilesN2) << 4;
  const int mlocal = lane & 15;
  const int halfSel = lane >> 4;                // K offset +8 for upper half-wave
  const int col0 = (tn << 5) + mlocal;
  const int col1 = col0 + 16;

  v8f acc0, acc1;
  const float b0 = bias ? bias[col0] : 0.0f;
  const float b1 = bias ? bias[col1] : 0.0f;
#pragma unroll
  for (int r = 0; r < 8; ++r) { acc0[r] = b0; acc1[r] = b1; }

  const AT* aRow = A + (long)(rowBase + mlocal) * K;
  const _Float16* bRow0 = Wh + (long)col0 * K;
  const _Float16* bRow1 = Wh + (long)col1 * K;

  for (int k0 = 0; k0 < K; k0 += 32) {
    const int kb = k0 + (halfSel << 3);
    v16h af  = pack_frag(aRow,  kb);
    v16h bf0 = pack_frag(bRow0, kb);
    v16h bf1 = pack_frag(bRow1, kb);
    acc0 = __builtin_amdgcn_wmma_f32_16x16x32_f16(false, af, false, bf0, (short)0, acc0, false, false);
    acc1 = __builtin_amdgcn_wmma_f32_16x16x32_f16(false, af, false, bf1, (short)0, acc1, false, false);
  }

#pragma unroll
  for (int r = 0; r < 8; ++r) {
    const long row = rowBase + r + halfSel * 8;
    float v0 = acc0[r], v1 = acc1[r];
    if (ACT == 1) { v0 = silu_f(v0); v1 = silu_f(v1); }
    if (RS == 1)  { const float cf = cutoff_fn(ew[row]); v0 *= cf; v1 *= cf; }
    C[row * (long)Nout + col0] = (OT)v0;
    C[row * (long)Nout + col1] = (OT)v1;
  }
}

// ---------------------------------------------------------------------------
// Concat-gather GEMM for Zij: A row e = concat(ZN[src[e]], ZN[dst[e]]), K = 256.
// ---------------------------------------------------------------------------
__global__ __launch_bounds__(256)
void k_gemm_concat(const _Float16* __restrict__ ZN,
                   const int* __restrict__ gsrc, const int* __restrict__ gdst,
                   const _Float16* __restrict__ Wh, const float* __restrict__ bias,
                   float* __restrict__ C, int M, int Nout)
{
  const int K = 2 * H_DIM;
  const int tilesN2 = Nout >> 5;
  const long total = (long)(M >> 4) * tilesN2;
  const int lane = threadIdx.x & 31;
  long t = (long)blockIdx.x * 8 + (threadIdx.x >> 5);
  if (t >= total) return;
  const int tn = (int)(t % tilesN2);
  const int rowBase = (int)(t / tilesN2) << 4;
  const int mlocal = lane & 15;
  const int halfSel = lane >> 4;
  const int col0 = (tn << 5) + mlocal;
  const int col1 = col0 + 16;
  const int e = rowBase + mlocal;
  const int sIdx = gsrc[e];
  const int dIdx = gdst[e];

  v8f acc0, acc1;
  const float b0 = bias[col0], b1 = bias[col1];
#pragma unroll
  for (int r = 0; r < 8; ++r) { acc0[r] = b0; acc1[r] = b1; }

  const _Float16* bRow0 = Wh + (long)col0 * K;
  const _Float16* bRow1 = Wh + (long)col1 * K;

#pragma unroll
  for (int half = 0; half < 2; ++half) {
    const _Float16* aRow = ZN + (long)(half ? dIdx : sIdx) * H_DIM;
    const int koff = half * H_DIM;
#pragma unroll
    for (int k0 = 0; k0 < H_DIM; k0 += 32) {
      const int kb = k0 + (halfSel << 3);
      v16h af  = pack_frag(aRow,  kb);
      v16h bf0 = pack_frag(bRow0, koff + kb);
      v16h bf1 = pack_frag(bRow1, koff + kb);
      acc0 = __builtin_amdgcn_wmma_f32_16x16x32_f16(false, af, false, bf0, (short)0, acc0, false, false);
      acc1 = __builtin_amdgcn_wmma_f32_16x16x32_f16(false, af, false, bf1, (short)0, acc1, false, false);
    }
  }

#pragma unroll
  for (int r = 0; r < 8; ++r) {
    const long row = rowBase + r + halfSel * 8;
    C[row * (long)Nout + col0] = acc0[r];
    C[row * (long)Nout + col1] = acc1[r];
  }
}

// ---------------------------------------------------------------------------
__global__ void k_zero(float* p, long n) {
  long i = (long)blockIdx.x * blockDim.x + threadIdx.x;
  if (i < n) p[i] = 0.0f;
}

__global__ void k_cvt_f16(const float* __restrict__ in, _Float16* __restrict__ out, long n) {
  long i = (long)blockIdx.x * blockDim.x + threadIdx.x;
  if (i < n) out[i] = (_Float16)in[i];
}

__global__ void k_embed_gather(const int* __restrict__ z, const float* __restrict__ emb,
                               _Float16* __restrict__ ZN) {
  long i = (long)blockIdx.x * blockDim.x + threadIdx.x;
  if (i >= (long)N_ATOMS * H_DIM) return;
  int n = (int)(i >> 7), h = (int)(i & 127);
  ZN[i] = (_Float16)emb[(long)z[n] * H_DIM + h];
}

// Embedding: scatter-add Zij * (W1*I + W2*skew(v) + W3*sym(v)) into compact planes
__global__ __launch_bounds__(128)
void k_edge_accum(const int* __restrict__ src, const float* __restrict__ vvec,
                  const float* __restrict__ zij, const float* __restrict__ w123,
                  float* __restrict__ planes) {
  const int e = blockIdx.x, h = threadIdx.x;
  const float vx = vvec[e*3+0], vy = vvec[e*3+1], vz = vvec[e*3+2];
  const long eh = (long)e * H_DIM + h;
  const long EH = (long)E_EDGES * H_DIM;
  const float zv = zij[eh];
  const float c1 = zv * w123[eh];
  const float c2 = zv * w123[EH + eh];
  const float c3 = zv * w123[2*EH + eh];
  const long NH = (long)N_ATOMS * H_DIM;
  const long base = (long)src[e] * H_DIM + h;
  atomicAdd(&planes[0*NH+base], c1);
  atomicAdd(&planes[1*NH+base], c2 * (-vz));
  atomicAdd(&planes[2*NH+base], c2 * ( vy));
  atomicAdd(&planes[3*NH+base], c2 * (-vx));
  atomicAdd(&planes[4*NH+base], c3 * (vx*vx - (1.0f/3.0f)));
  atomicAdd(&planes[5*NH+base], c3 * (vy*vy - (1.0f/3.0f)));
  atomicAdd(&planes[6*NH+base], c3 * (vz*vz - (1.0f/3.0f)));
  atomicAdd(&planes[7*NH+base], c3 * (vx*vy));
  atomicAdd(&planes[8*NH+base], c3 * (vx*vz));
  atomicAdd(&planes[9*NH+base], c3 * (vy*vz));
}

// tensor_norm(I+A+S) then LayerNorm over H (block per atom); f16 out (GEMM-only consumer)
__global__ __launch_bounds__(128)
void k_norm_ln(const float* __restrict__ planes, const float* __restrict__ g,
               const float* __restrict__ bb, _Float16* __restrict__ out) {
  const int n = blockIdx.x, h = threadIdx.x;
  const long NH = (long)N_ATOMS * H_DIM;
  const long idx = (long)n * H_DIM + h;
  float c[10];
#pragma unroll
  for (int p = 0; p < 10; ++p) c[p] = planes[p*NH + idx];
  float T[9]; full_from_compact(c, T);
  float x = 0.0f;
#pragma unroll
  for (int k = 0; k < 9; ++k) x += T[k]*T[k];
  __shared__ float sm[128];
  sm[h] = x; __syncthreads();
  for (int s = 64; s > 0; s >>= 1) { if (h < s) sm[h] += sm[h+s]; __syncthreads(); }
  const float mean = sm[0] * (1.0f/128.0f); __syncthreads();
  const float d = x - mean;
  sm[h] = d*d; __syncthreads();
  for (int s = 64; s > 0; s >>= 1) { if (h < s) sm[h] += sm[h+s]; __syncthreads(); }
  const float var = sm[0] * (1.0f/128.0f);
  out[idx] = (_Float16)(d * rsqrtf(var + 1e-5f) * g[h] + bb[h]);
}

// X = f0*I' + f1*A' + f2*S'  ->  d_out [N,H,3,3]
__global__ void k_apply_embed(const float* __restrict__ planes, const float* __restrict__ nrm3,
                              float* __restrict__ X) {
  long i = (long)blockIdx.x * blockDim.x + threadIdx.x;
  const long NH = (long)N_ATOMS * H_DIM;
  if (i >= NH) return;
  const int n = (int)(i >> 7), h = (int)(i & 127);
  float c[10];
#pragma unroll
  for (int p = 0; p < 10; ++p) c[p] = planes[p*NH + i];
  const float f0 = nrm3[(long)n*384 + h*3 + 0];
  const float f1 = nrm3[(long)n*384 + h*3 + 1];
  const float f2 = nrm3[(long)n*384 + h*3 + 2];
  float s[10] = { f0*c[0], f1*c[1], f1*c[2], f1*c[3],
                  f2*c[4], f2*c[5], f2*c[6], f2*c[7], f2*c[8], f2*c[9] };
  float T[9]; full_from_compact(s, T);
  float* o = X + i*9;
#pragma unroll
  for (int k = 0; k < 9; ++k) o[k] = T[k];
}

// Interaction: normalize X in place, emit f16 compact decomposition (GEMM-only consumer)
__global__ void k_prep(float* __restrict__ X, _Float16* __restrict__ planes) {
  long i = (long)blockIdx.x * blockDim.x + threadIdx.x;
  const long NH = (long)N_ATOMS * H_DIM;
  if (i >= NH) return;
  float* x = X + i*9;
  float T[9], nrm = 0.0f;
#pragma unroll
  for (int k = 0; k < 9; ++k) { T[k] = x[k]; nrm += T[k]*T[k]; }
  const float inv = 1.0f / (nrm + 1.0f);
#pragma unroll
  for (int k = 0; k < 9; ++k) { T[k] *= inv; x[k] = T[k]; }
  float c[10]; compact_from_full(T, c);
#pragma unroll
  for (int p = 0; p < 10; ++p) planes[p*NH + i] = (_Float16)c[p];
}

// message passing: gather Y-compact at dst, scale by ea components, scatter to src
__global__ __launch_bounds__(128)
void k_msg(const int* __restrict__ src, const int* __restrict__ dst,
           const float* __restrict__ ea3, const float* __restrict__ planesY,
           float* __restrict__ msg) {
  const int e = blockIdx.x, h = threadIdx.x;
  const long NH = (long)N_ATOMS * H_DIM;
  const long gi = (long)dst[e] * H_DIM + h;
  const long si = (long)src[e] * H_DIM + h;
  const float e0 = ea3[(long)e*384 + h*3 + 0];
  const float e1 = ea3[(long)e*384 + h*3 + 1];
  const float e2 = ea3[(long)e*384 + h*3 + 2];
  atomicAdd(&msg[0*NH+si], e0 * planesY[0*NH+gi]);
#pragma unroll
  for (int p = 1; p < 4; ++p)  atomicAdd(&msg[p*NH+si], e1 * planesY[p*NH+gi]);
#pragma unroll
  for (int p = 4; p < 10; ++p) atomicAdd(&msg[p*NH+si], e2 * planesY[p*NH+gi]);
}

// T = M@Y + Y@M; compact(T)/(||T||^2+1) -> f16 planes (GEMM-only consumer)
__global__ void k_group(const float* __restrict__ planesY, const float* __restrict__ msg,
                        _Float16* __restrict__ planesOut) {
  long i = (long)blockIdx.x * blockDim.x + threadIdx.x;
  const long NH = (long)N_ATOMS * H_DIM;
  if (i >= NH) return;
  float cy[10], cm[10];
#pragma unroll
  for (int p = 0; p < 10; ++p) { cy[p] = planesY[p*NH + i]; cm[p] = msg[p*NH + i]; }
  float Y[9], M[9], A2[9], B2[9], T[9];
  full_from_compact(cy, Y); full_from_compact(cm, M);
  mm3(M, Y, A2); mm3(Y, M, B2);
  float ss = 0.0f;
#pragma unroll
  for (int k = 0; k < 9; ++k) { T[k] = A2[k] + B2[k]; ss += T[k]*T[k]; }
  const float inv = 1.0f / (ss + 1.0f);
  float c[10]; compact_from_full(T, c);
#pragma unroll
  for (int p = 0; p < 10; ++p) planesOut[p*NH + i] = (_Float16)(c[p] * inv);
}

// X = X + dX + dX@dX
__global__ void k_finish(const float* __restrict__ planesD, float* __restrict__ X) {
  long i = (long)blockIdx.x * blockDim.x + threadIdx.x;
  const long NH = (long)N_ATOMS * H_DIM;
  if (i >= NH) return;
  float c[10];
#pragma unroll
  for (int p = 0; p < 10; ++p) c[p] = planesD[p*NH + i];
  float D[9], DD[9]; full_from_compact(c, D); mm3(D, D, DD);
  float* x = X + i*9;
#pragma unroll
  for (int k = 0; k < 9; ++k) x[k] = x[k] + D[k] + DD[k];
}

// ---------------------------------------------------------------------------
template<typename AT, typename OT, int ACT, int RS>
static inline void launch_gemm(const AT* A, const _Float16* Wh, const float* bias, OT* C,
                               int M, int Nout, int K, const float* ew, hipStream_t s) {
  long tiles = ((long)M / 16) * (Nout / 32);
  int blocks = (int)((tiles + 7) / 8);
  k_gemm<AT, OT, ACT, RS><<<blocks, 256, 0, s>>>(A, Wh, bias, C, M, Nout, K, ew);
}

extern "C" void kernel_launch(void* const* d_in, const int* in_sizes, int n_in,
                              void* d_out, int out_size, void* d_ws, size_t ws_size,
                              hipStream_t stream) {
  (void)in_sizes; (void)n_in; (void)out_size; (void)ws_size;
  const int*   z     = (const int*)d_in[0];
  const int*   eidx  = (const int*)d_in[1];
  const int*   src   = eidx;
  const int*   dst   = eidx + E_EDGES;
  const float* ew    = (const float*)d_in[2];
  const float* ev    = (const float*)d_in[3];
  const float* eattr = (const float*)d_in[4];
  // params pytree, sorted-key flatten: dp1,dp2,dp3, emb, emb2, emb_ls, emb_lt, layers, ln
  const float* dp1W = (const float*)d_in[5];  const float* dp1b = (const float*)d_in[6];
  const float* dp2W = (const float*)d_in[7];  const float* dp2b = (const float*)d_in[8];
  const float* dp3W = (const float*)d_in[9];  const float* dp3b = (const float*)d_in[10];
  const float* emb  = (const float*)d_in[11];
  const float* e2W  = (const float*)d_in[12]; const float* e2b  = (const float*)d_in[13];
  const float* ls0W = (const float*)d_in[14]; const float* ls0b = (const float*)d_in[15];
  const float* ls1W = (const float*)d_in[16]; const float* ls1b = (const float*)d_in[17];
  const float* eltW[3] = { (const float*)d_in[18], (const float*)d_in[19], (const float*)d_in[20] };
  const float* lnB  = (const float*)d_in[45]; const float* lnG  = (const float*)d_in[46];

  const long NH = (long)N_ATOMS * H_DIM;
  const long EH = (long)E_EDGES * H_DIM;
  float* ws      = (float*)d_ws;
  float*     planesA  = ws;                 // 10*NH f32 (embedding atomic accumulators)
  _Float16*  planesAh = (_Float16*)ws;      // same slab reused as f16 in interactions
  float*     planesB  = ws + 10*NH;         // 10*NH f32 (post-chan_lin)
  float*     msgp     = ws + 20*NH;         // 10*NH f32 (messages; aliased LN-MLP bufs)
  _Float16*  nrm1h    = (_Float16*)msgp;            // NH halves
  _Float16*  nrm2h    = (_Float16*)(msgp + NH);     // 2*NH halves
  float*     nrm3     = msgp + 3*NH;                // 3*NH f32
  _Float16*  ZNh      = (_Float16*)(ws + 30*NH);    // NH halves
  _Float16*  ea1h     = (_Float16*)(ws + 31*NH);    // EH halves   |  (within 2*EH slab)
  _Float16*  ea2h     = (_Float16*)(ws + 31*NH + EH); // 2*EH halves
  float*     zijbuf   = ws + 31*NH + 2*EH;  // EH f32 (Zij)
  float*     w123     = ws + 31*NH + 3*EH;  // 3*EH f32 (W1W2W3 | ea3)
  _Float16*  wcache   = (_Float16*)(ws + 31*NH + 6*EH); // f16 weight cache (~1.4 MB)
  float*     Xout     = (float*)d_out;

  const int EW = 256;
  auto blocksFor = [&](long n) { return (int)((n + EW - 1) / EW); };

  // ---- pre-convert all GEMM weights to f16 once ----
  size_t wcOff = 0;
  auto cvtW = [&](const float* w, long n) -> _Float16* {
    _Float16* o = wcache + wcOff;
    wcOff += (size_t)((n + 7) & ~7L);     // keep 16B alignment
    k_cvt_f16<<<blocksFor(n), EW, 0, stream>>>(w, o, n);
    return o;
  };
  _Float16* dp1Wh = cvtW(dp1W, H_DIM*R_RBF);
  _Float16* dp2Wh = cvtW(dp2W, H_DIM*R_RBF);
  _Float16* dp3Wh = cvtW(dp3W, H_DIM*R_RBF);
  _Float16* e2Wh  = cvtW(e2W,  H_DIM*2*H_DIM);
  _Float16* ls0Wh = cvtW(ls0W, 2*H_DIM*H_DIM);
  _Float16* ls1Wh = cvtW(ls1W, 3*H_DIM*2*H_DIM);
  _Float16* eltWh[3];
  for (int j = 0; j < 3; ++j) eltWh[j] = cvtW(eltW[j], H_DIM*H_DIM);
  _Float16* LsWh[2][3]; const float* Lsb[2][3]; _Float16* ltWh[2][6];
  for (int l = 0; l < 2; ++l) {
    const int base = 21 + l * 12;
    LsWh[l][0] = cvtW((const float*)d_in[base+0],   H_DIM*R_RBF);   Lsb[l][0] = (const float*)d_in[base+1];
    LsWh[l][1] = cvtW((const float*)d_in[base+2], 2*H_DIM*H_DIM);   Lsb[l][1] = (const float*)d_in[base+3];
    LsWh[l][2] = cvtW((const float*)d_in[base+4], 3*H_DIM*2*H_DIM); Lsb[l][2] = (const float*)d_in[base+5];
    for (int j = 0; j < 6; ++j) ltWh[l][j] = cvtW((const float*)d_in[base+6+j], H_DIM*H_DIM);
  }

  // ---------------- Embedding ----------------
  k_zero<<<blocksFor(10*NH), EW, 0, stream>>>(planesA, 10*NH);
  k_embed_gather<<<blocksFor(NH), EW, 0, stream>>>(z, emb, ZNh);
  launch_gemm<float,float,0,1>(eattr, dp1Wh, dp1b, w123 + 0*EH, E_EDGES, H_DIM, R_RBF, ew, stream);
  launch_gemm<float,float,0,1>(eattr, dp2Wh, dp2b, w123 + 1*EH, E_EDGES, H_DIM, R_RBF, ew, stream);
  launch_gemm<float,float,0,1>(eattr, dp3Wh, dp3b, w123 + 2*EH, E_EDGES, H_DIM, R_RBF, ew, stream);
  {
    long tiles = ((long)E_EDGES / 16) * (H_DIM / 32);
    int blocks = (int)((tiles + 7) / 8);
    k_gemm_concat<<<blocks, 256, 0, stream>>>(ZNh, src, dst, e2Wh, e2b, zijbuf, E_EDGES, H_DIM);
  }
  k_edge_accum<<<E_EDGES, H_DIM, 0, stream>>>(src, ev, zijbuf, w123, planesA);
  k_norm_ln<<<N_ATOMS, H_DIM, 0, stream>>>(planesA, lnG, lnB, nrm1h);
  launch_gemm<_Float16,_Float16,1,0>(nrm1h, ls0Wh, ls0b, nrm2h, N_ATOMS, 2*H_DIM, H_DIM, ew, stream);
  launch_gemm<_Float16,float,1,0>(nrm2h, ls1Wh, ls1b, nrm3, N_ATOMS, 3*H_DIM, 2*H_DIM, ew, stream);
  for (int p = 0; p < 10; ++p) {
    _Float16* lw = (p == 0) ? eltWh[0] : (p < 4 ? eltWh[1] : eltWh[2]);
    launch_gemm<float,float,0,0>(planesA + p*NH, lw, nullptr, planesB + p*NH,
                                 N_ATOMS, H_DIM, H_DIM, ew, stream);
  }
  k_apply_embed<<<blocksFor(NH), EW, 0, stream>>>(planesB, nrm3, Xout);

  // ---------------- Interaction layers ----------------
  for (int l = 0; l < 2; ++l) {
    launch_gemm<float,_Float16,1,0>(eattr, LsWh[l][0], Lsb[l][0], ea1h, E_EDGES,   H_DIM, R_RBF,   ew, stream);
    launch_gemm<_Float16,_Float16,1,0>(ea1h, LsWh[l][1], Lsb[l][1], ea2h, E_EDGES, 2*H_DIM, H_DIM,   ew, stream);
    launch_gemm<_Float16,float,1,1>(ea2h, LsWh[l][2], Lsb[l][2], w123, E_EDGES, 3*H_DIM, 2*H_DIM, ew, stream);

    k_prep<<<blocksFor(NH), EW, 0, stream>>>(Xout, planesAh);
    for (int p = 0; p < 10; ++p) {
      _Float16* lw = (p == 0) ? ltWh[l][0] : (p < 4 ? ltWh[l][1] : ltWh[l][2]);
      launch_gemm<_Float16,float,0,0>(planesAh + p*NH, lw, nullptr, planesB + p*NH,
                                      N_ATOMS, H_DIM, H_DIM, ew, stream);
    }
    k_zero<<<blocksFor(10*NH), EW, 0, stream>>>(msgp, 10*NH);
    k_msg<<<E_EDGES, H_DIM, 0, stream>>>(src, dst, w123, planesB, msgp);
    k_group<<<blocksFor(NH), EW, 0, stream>>>(planesB, msgp, planesAh);
    for (int p = 0; p < 10; ++p) {
      _Float16* lw = (p == 0) ? ltWh[l][3] : (p < 4 ? ltWh[l][4] : ltWh[l][5]);
      launch_gemm<_Float16,float,0,0>(planesAh + p*NH, lw, nullptr, planesB + p*NH,
                                      N_ATOMS, H_DIM, H_DIM, ew, stream);
    }
    k_finish<<<blocksFor(NH), EW, 0, stream>>>(planesB, Xout);
  }
}
